// RelationAwareGraphSelfAttention_78855599555043
// MI455X (gfx1250) — matrix operand order repaired and checked by reading, not compiled, not run on hardware
//
#include <hip/hip_runtime.h>
#include <math.h>

typedef __attribute__((ext_vector_type(16))) _Float16 v16h;
typedef __attribute__((ext_vector_type(8)))  float    v8f;

#define BATCH 2
#define NTOK  1024
#define QD    512
#define NH    4
#define DH    64
#define HD    64
#define CD    16
#define INR   256
#define LN_EPS 1e-5f
#define BIG_F  3.402823466e38f

// padded LDS strides (halves) to keep 16B alignment and spread banks
#define W1S_STR 40    // w1 tile [64 n][32 k] -> 80B rows
#define W2S_STR 72    // w2 tile [64 n][64 k] -> 144B rows
#define BT_STR  40    // generic B tile [64 n][32 k]
#define AT16_STR 1032 // attn f16 [16][1024] -> 2064B rows

// ---------------- WMMA fragment index helpers (contiguous k order, same for A and B) ----------
// A (16x32 f16): lane L holds row M=L&15; K = 16*(L>=16) + hh  (16 contiguous halves)
__device__ __forceinline__ int a_kidx(int hh, int lane) {
  return (((lane >> 4) & 1) << 4) + hh;
}
// B (32x16 f16): lane L holds col N=L&15; K = 16*(L>=16) + hh  (16 contiguous halves)
__device__ __forceinline__ int b_kidx(int hh, int lane) {
  return (((lane >> 4) & 1) << 4) + hh;
}
// C/D (16x16 f32): VGPR r -> M = r + 8*(L>=16), N = L&15

__device__ __forceinline__ v8f zero8() {
  v8f z;
#pragma unroll
  for (int i = 0; i < 8; ++i) z[i] = 0.0f;
  return z;
}

__device__ __forceinline__ v8f wmma32(v16h a, v16h b, v8f c) {
  return __builtin_amdgcn_wmma_f32_16x16x32_f16(
      /*neg_a=*/false, a, /*neg_b=*/false, b,
      /*c_mod=*/(short)0, c, /*reuse_a=*/false, /*reuse_b=*/false);
}

__device__ __forceinline__ float gelu_exact(float x) {
  return 0.5f * x * (1.0f + erff(x * 0.70710678118654752440f));
}

// =============================================================================
// Kernel 1: pairwise structural bias.
// Grid (64,64,2); 256 threads = 8 waves; 64KB dynamic LDS (phased reuse).
// =============================================================================
__global__ void __launch_bounds__(256)
pair_bias_kernel(const float* __restrict__ cf,
                 const unsigned char* __restrict__ pair_mask,
                 const int* __restrict__ db, const int* __restrict__ dirb,
                 const int* __restrict__ roleb, const int* __restrict__ hopb,
                 const int* __restrict__ edgeb, const int* __restrict__ spb,
                 const int* __restrict__ sameb,
                 const int* __restrict__ degids, const int* __restrict__ ctids,
                 const float* __restrict__ w1, const float* __restrict__ b1,
                 const float* __restrict__ w2, const float* __restrict__ b2,
                 const float* __restrict__ dist_e, const float* __restrict__ dir_e,
                 const float* __restrict__ role_e, const float* __restrict__ hop_e,
                 const float* __restrict__ edge_e, const float* __restrict__ sp_e,
                 const float* __restrict__ deg_e, const float* __restrict__ ct_e,
                 const float* __restrict__ same_e,
                 const float* __restrict__ ln_g, const float* __restrict__ ln_b,
                 const float* __restrict__ out_w, const float* __restrict__ out_b,
                 float* __restrict__ bias_out)
{
  extern __shared__ char smem[];
  const int bb = blockIdx.z;
  const int i0 = blockIdx.x << 4;
  const int j0 = blockIdx.y << 4;
  const int tid = threadIdx.x, lane = tid & 31, wave = tid >> 5;

  // ---- phase 1: A = cf tile (f16, K padded 16->32); B = w1 transposed ------
  _Float16* abuf = (_Float16*)smem;               // [256][32]       16KB @0
  _Float16* w1s  = (_Float16*)(smem + 16 * 1024); // [64 n][W1S_STR]  5KB @16K
  {
    const int p = tid;
    const int i = i0 + (p >> 4), j = j0 + (p & 15);
    const float* src = cf + (((size_t)bb * NTOK + i) * NTOK + j) * CD;
#pragma unroll
    for (int k = 0; k < CD; ++k) abuf[p * 32 + k] = (_Float16)src[k];
#pragma unroll
    for (int k = CD; k < 32; ++k) abuf[p * 32 + k] = (_Float16)0.0f;
  }
  for (int e = tid; e < 32 * HD; e += 256) {
    const int k = e >> 6, n = e & 63;        // consecutive lanes -> consecutive n
    w1s[n * W1S_STR + k] = (k < CD) ? (_Float16)w1[k * HD + n] : (_Float16)0.0f;
  }
  __syncthreads();

  // ---- layer 1: [256,32] x [32,64], each wave owns 2 m-tiles x 4 n-tiles ---
  v8f c1[2][4];
#pragma unroll
  for (int t = 0; t < 2; ++t)
#pragma unroll
    for (int nt = 0; nt < 4; ++nt) c1[t][nt] = zero8();
  {
    v16h A[2], Bm[4];
#pragma unroll
    for (int t = 0; t < 2; ++t) {
      const int row = (wave * 2 + t) * 16 + (lane & 15);
#pragma unroll
      for (int hh = 0; hh < 16; ++hh) A[t][hh] = abuf[row * 32 + a_kidx(hh, lane)];
    }
#pragma unroll
    for (int nt = 0; nt < 4; ++nt) {
      const int n = nt * 16 + (lane & 15);
#pragma unroll
      for (int hh = 0; hh < 16; ++hh)
        Bm[nt][hh] = w1s[n * W1S_STR + b_kidx(hh, lane)];
    }
#pragma unroll
    for (int t = 0; t < 2; ++t)
#pragma unroll
      for (int nt = 0; nt < 4; ++nt)
        c1[t][nt] = wmma32(A[t], Bm[nt], c1[t][nt]);
  }
  __syncthreads();

  // ---- phase 2: hid = gelu(c1 + b1) as f16; w2 transposed ------------------
  _Float16* hid = (_Float16*)smem;               // [256][64]       32KB @0
  _Float16* w2s = (_Float16*)(smem + 32 * 1024); // [64 n][W2S_STR]  9KB @32K
  float b1v[4];
#pragma unroll
  for (int nt = 0; nt < 4; ++nt) b1v[nt] = b1[nt * 16 + (lane & 15)];
#pragma unroll
  for (int t = 0; t < 2; ++t)
#pragma unroll
    for (int nt = 0; nt < 4; ++nt) {
#pragma unroll
      for (int r = 0; r < 8; ++r) {
        const int row = (wave * 2 + t) * 16 + r + ((lane >> 4) << 3);
        const int col = nt * 16 + (lane & 15);
        hid[row * HD + col] = (_Float16)gelu_exact(c1[t][nt][r] + b1v[nt]);
      }
    }
  for (int e = tid; e < HD * HD; e += 256) {
    const int k = e >> 6, n = e & 63;
    w2s[n * W2S_STR + k] = (_Float16)w2[k * HD + n];
  }
  __syncthreads();

  // ---- layer 2: [256,64] x [64,64] (K = 2 x 32) ----------------------------
  v8f c2[2][4];
#pragma unroll
  for (int t = 0; t < 2; ++t)
#pragma unroll
    for (int nt = 0; nt < 4; ++nt) c2[t][nt] = zero8();
#pragma unroll
  for (int ks = 0; ks < 2; ++ks) {
    v16h A[2], Bm[4];
#pragma unroll
    for (int t = 0; t < 2; ++t) {
      const int row = (wave * 2 + t) * 16 + (lane & 15);
#pragma unroll
      for (int hh = 0; hh < 16; ++hh)
        A[t][hh] = hid[row * HD + ks * 32 + a_kidx(hh, lane)];
    }
#pragma unroll
    for (int nt = 0; nt < 4; ++nt) {
      const int n = nt * 16 + (lane & 15);
#pragma unroll
      for (int hh = 0; hh < 16; ++hh)
        Bm[nt][hh] = w2s[n * W2S_STR + ks * 32 + b_kidx(hh, lane)];
    }
#pragma unroll
    for (int t = 0; t < 2; ++t)
#pragma unroll
      for (int nt = 0; nt < 4; ++nt)
        c2[t][nt] = wmma32(A[t], Bm[nt], c2[t][nt]);
  }
  __syncthreads();

  // ---- phase 3: layer-2 result to f32 pairbuf (LDS) ------------------------
  float* pairbuf = (float*)smem;                 // [256][64] 64KB @0
#pragma unroll
  for (int t = 0; t < 2; ++t)
#pragma unroll
    for (int nt = 0; nt < 4; ++nt) {
#pragma unroll
      for (int r = 0; r < 8; ++r) {
        const int row = (wave * 2 + t) * 16 + r + ((lane >> 4) << 3);
        const int col = nt * 16 + (lane & 15);
        pairbuf[row * HD + col] = c2[t][nt][r];
      }
    }
  __syncthreads();

  // ---- epilogue: one pair per thread ---------------------------------------
  {
    const int p = tid;
    const int i = i0 + (p >> 4), j = j0 + (p & 15);
    const size_t pij = ((size_t)bb * NTOK + i) * NTOK + j;
    const int e0 = db[pij] * HD,    e1 = dirb[pij] * HD;
    const int e2 = roleb[pij] * HD, e3 = hopb[pij] * HD;
    const int e4 = edgeb[pij] * HD, e5 = spb[pij] * HD;
    const int e6 = sameb[pij] * HD;
    const int d0 = degids[bb * NTOK + i] * HD, d1 = degids[bb * NTOK + j] * HD;
    const int t0 = ctids[bb * NTOK + i] * HD,  t1 = ctids[bb * NTOK + j] * HD;

    float mean = 0.0f, msq = 0.0f;
    for (int c = 0; c < HD; ++c) {
      float v = pairbuf[p * HD + c] + b2[c]
              + dist_e[e0 + c] + dir_e[e1 + c] + role_e[e2 + c] + hop_e[e3 + c]
              + edge_e[e4 + c] + sp_e[e5 + c] + same_e[e6 + c]
              + deg_e[d0 + c] + deg_e[d1 + c] + ct_e[t0 + c] + ct_e[t1 + c];
      pairbuf[p * HD + c] = v;
      mean += v;
      msq  += v * v;
    }
    mean *= (1.0f / HD);
    const float var  = msq * (1.0f / HD) - mean * mean;
    const float rstd = rsqrtf(var + LN_EPS);
    float a0 = out_b[0], a1 = out_b[1], a2 = out_b[2], a3 = out_b[3];
    for (int c = 0; c < HD; ++c) {
      const float lnv = (pairbuf[p * HD + c] - mean) * rstd * ln_g[c] + ln_b[c];
      const float g = gelu_exact(lnv);
      a0 += g * out_w[c * NH + 0];
      a1 += g * out_w[c * NH + 1];
      a2 += g * out_w[c * NH + 2];
      a3 += g * out_w[c * NH + 3];
    }
    const float m = pair_mask[pij] ? 1.0f : 0.0f;
    const size_t ob = (((size_t)bb * NH) * NTOK + i) * NTOK + j;
    const size_t hstride = (size_t)NTOK * NTOK;
    bias_out[ob + 0 * hstride] = a0 * m;
    bias_out[ob + 1 * hstride] = a1 * m;
    bias_out[ob + 2 * hstride] = a2 * m;
    bias_out[ob + 3 * hstride] = a3 * m;
  }
}

// =============================================================================
// Kernel 2: q/k/v projections, x[2048,512] x W[512,256] -> f16 workspace.
// Grid (32,4,3); 256 threads; ~9.25KB LDS.
// =============================================================================
__global__ void __launch_bounds__(256)
qkv_kernel(const float* __restrict__ x,
           const float* __restrict__ Wq, const float* __restrict__ Wk,
           const float* __restrict__ Wv, _Float16* __restrict__ qkv)
{
  extern __shared__ char smem[];
  _Float16* at = (_Float16*)smem;            // [64][32]        4KB
  _Float16* bt = (_Float16*)(smem + 4096);   // [64 n][BT_STR]  5KB (transposed)
  const int which = blockIdx.z;
  const float* W = (which == 0) ? Wq : (which == 1) ? Wk : Wv;
  _Float16* dst = qkv + (size_t)which * (BATCH * NTOK) * INR;
  const int m0 = blockIdx.x << 6, n0 = blockIdx.y << 6;
  const int tid = threadIdx.x, lane = tid & 31, wave = tid >> 5;

  v8f c0 = zero8(), c1 = zero8();
  for (int k0 = 0; k0 < QD; k0 += 32) {
    for (int e = tid; e < 64 * 32; e += 256) {
      const int r = e >> 5, kk = e & 31;
      at[e] = (_Float16)x[(size_t)(m0 + r) * QD + k0 + kk];
    }
    for (int e = tid; e < 32 * 64; e += 256) {
      const int kk = e >> 6, n = e & 63;     // consecutive lanes -> consecutive n
      bt[n * BT_STR + kk] = (_Float16)W[(size_t)(k0 + kk) * INR + n0 + n];
    }
    __syncthreads();
#pragma unroll
    for (int t = 0; t < 2; ++t) {
      const int s = wave * 2 + t, mt = s >> 2, nt = s & 3;
      v16h a, bm;
      const int row = mt * 16 + (lane & 15);
#pragma unroll
      for (int hh = 0; hh < 16; ++hh) a[hh] = at[row * 32 + a_kidx(hh, lane)];
      const int n = nt * 16 + (lane & 15);
#pragma unroll
      for (int hh = 0; hh < 16; ++hh) bm[hh] = bt[n * BT_STR + b_kidx(hh, lane)];
      if (t == 0) c0 = wmma32(a, bm, c0);
      else        c1 = wmma32(a, bm, c1);
    }
    __syncthreads();
  }
#pragma unroll
  for (int t = 0; t < 2; ++t) {
    const int s = wave * 2 + t, mt = s >> 2, nt = s & 3;
    const v8f cc = (t == 0) ? c0 : c1;
#pragma unroll
    for (int r = 0; r < 8; ++r) {
      const int row = m0 + mt * 16 + r + ((lane >> 4) << 3);
      const int col = n0 + nt * 16 + (lane & 15);
      dst[(size_t)row * INR + col] = (_Float16)cc[r];
    }
  }
}

// =============================================================================
// Kernel 3: attention. One WG per (b, head, 16-row i tile).
// Grid (64,4,2); 256 threads; ~108KB dynamic LDS.
// =============================================================================
__global__ void __launch_bounds__(256)
attn_kernel(const _Float16* __restrict__ qkv,
            const float* __restrict__ bias,           // d_out bias region [B,H,N,N]
            const unsigned char* __restrict__ pair_mask,
            const unsigned char* __restrict__ valid_mask,
            float* __restrict__ attn_out,             // d_out attn region [B,H,N,N]
            float* __restrict__ ctx)                  // [B*N, INR] f32
{
  extern __shared__ char smem[];
  float*    logits = (float*)smem;                          // [16][1024] 64KB
  _Float16* qtile  = (_Float16*)(smem + 65536);             // [16][64]    2KB
  float*    red    = (float*)(smem + 65536 + 2048);         // [16][16]    1KB
  float*    outacc = (float*)(smem + 65536 + 2048 + 1024);  // [2][16][64] 8KB
  _Float16* att16  = (_Float16*)(smem + 65536 + 2048 + 1024 + 8192); // [16][AT16_STR]

  const int bb = blockIdx.z, hh0 = blockIdx.y, i0 = blockIdx.x << 4;
  const int tid = threadIdx.x, lane = tid & 31, wave = tid >> 5;
  const _Float16* q    = qkv;
  const _Float16* kmat = qkv + (size_t)(BATCH * NTOK) * INR;
  const _Float16* vmat = qkv + (size_t)2 * (BATCH * NTOK) * INR;

  for (int e = tid; e < 16 * DH; e += 256) {
    const int r = e >> 6, d = e & 63;
    qtile[e] = q[((size_t)(bb * NTOK) + i0 + r) * INR + hh0 * DH + d];
  }
  __syncthreads();

  // q A-fragments (K=64 -> 2 steps), reused for all j tiles of this wave
  v16h Aq0, Aq1;
  {
    const int row = lane & 15;
#pragma unroll
    for (int t = 0; t < 16; ++t) Aq0[t] = qtile[row * DH + a_kidx(t, lane)];
#pragma unroll
    for (int t = 0; t < 16; ++t) Aq1[t] = qtile[row * DH + 32 + a_kidx(t, lane)];
  }

  const float scale = 0.125f;  // 1/sqrt(64)
  for (int jt = wave; jt < NTOK / 16; jt += 8) {
    v8f c = zero8();
    const int j = jt * 16 + (lane & 15);
    {
      v16h bm0, bm1;  // k rows are contiguous 32B per lane -> vector loads
#pragma unroll
      for (int t = 0; t < 16; ++t)
        bm0[t] = kmat[((size_t)(bb * NTOK) + j) * INR + hh0 * DH + b_kidx(t, lane)];
#pragma unroll
      for (int t = 0; t < 16; ++t)
        bm1[t] = kmat[((size_t)(bb * NTOK) + j) * INR + hh0 * DH + 32 + b_kidx(t, lane)];
      c = wmma32(Aq0, bm0, c);
      c = wmma32(Aq1, bm1, c);
    }
#pragma unroll
    for (int r = 0; r < 8; ++r) {
      const int i = i0 + r + ((lane >> 4) << 3);
      const size_t pij = ((size_t)bb * NTOK + i) * NTOK + j;
      const float val = c[r] * scale +
          bias[(((size_t)bb * NH + hh0) * NTOK + i) * NTOK + j];
      const bool safe =
          pair_mask[pij] || ((!valid_mask[bb * NTOK + i]) && (i == j));
      logits[(i - i0) * NTOK + j] = safe ? val : -BIG_F;
    }
  }
  __syncthreads();

  // ---- softmax + mask renorm: 16 threads/row, interleaved j (bank friendly)
  const int r = tid >> 4, seg = tid & 15;
  float* Lrow = logits + r * NTOK;

  float mx = -BIG_F;
  for (int jj = 0; jj < 64; ++jj) mx = fmaxf(mx, Lrow[seg + jj * 16]);
  red[r * 16 + seg] = mx;
  __syncthreads();
  float rowmax = red[r * 16];
  for (int s2 = 1; s2 < 16; ++s2) rowmax = fmaxf(rowmax, red[r * 16 + s2]);
  __syncthreads();

  float ps = 0.0f;
  for (int jj = 0; jj < 64; ++jj) {
    const int j = seg + jj * 16;
    const float p = expf(Lrow[j] - rowmax);
    Lrow[j] = p;
    ps += p;
  }
  red[r * 16 + seg] = ps;
  __syncthreads();
  float S = 0.0f;
  for (int s2 = 0; s2 < 16; ++s2) S += red[r * 16 + s2];
  __syncthreads();

  const float invS = 1.0f / S;
  float ms = 0.0f;
  {
    const size_t base = ((size_t)bb * NTOK + (i0 + r)) * NTOK;
    for (int jj = 0; jj < 64; ++jj) {
      const int j = seg + jj * 16;
      const float mval = pair_mask[base + j] ? 1.0f : 0.0f;
      const float qv = Lrow[j] * invS * mval;
      Lrow[j] = qv;
      ms += qv;
    }
  }
  red[r * 16 + seg] = ms;
  __syncthreads();
  float S2 = 0.0f;
  for (int s2 = 0; s2 < 16; ++s2) S2 += red[r * 16 + s2];
  const float invD = 1.0f / fmaxf(S2, 1e-6f);
  {
    const size_t obase = (((size_t)bb * NH + hh0) * NTOK + (i0 + r)) * NTOK;
    for (int jj = 0; jj < 64; ++jj) {
      const int j = seg + jj * 16;
      const float a = Lrow[j] * invD;
      att16[r * AT16_STR + j] = (_Float16)a;   // f16 copy for the PV WMMA
      attn_out[obase + j] = a;                 // coalesced within 16-lane groups
    }
  }
  __syncthreads();

  // ---- ctx = attn @ v : waves split (n-subtile, K-half) --------------------
  const int nt = wave & 3, kh = wave >> 2;
  v8f co = zero8();
  for (int kstep = kh * 16; kstep < kh * 16 + 16; ++kstep) {
    v16h a, bm;
    const int row = lane & 15;
#pragma unroll
    for (int t = 0; t < 16; ++t)
      a[t] = att16[row * AT16_STR + kstep * 32 + a_kidx(t, lane)];
    const int d = nt * 16 + (lane & 15);
#pragma unroll
    for (int t = 0; t < 16; ++t) {
      const int j = kstep * 32 + b_kidx(t, lane);
      bm[t] = vmat[((size_t)(bb * NTOK) + j) * INR + hh0 * DH + d];
    }
    co = wmma32(a, bm, co);
  }
#pragma unroll
  for (int r2 = 0; r2 < 8; ++r2) {
    const int row = r2 + ((lane >> 4) << 3);
    const int col = nt * 16 + (lane & 15);
    outacc[kh * (16 * DH) + row * DH + col] = co[r2];
  }
  __syncthreads();
  for (int e = tid; e < 16 * DH; e += 256) {
    const int rr = e >> 6, d = e & 63;
    ctx[((size_t)(bb * NTOK) + i0 + rr) * INR + hh0 * DH + d] =
        outacc[e] + outacc[16 * DH + e];
  }
}

// =============================================================================
// Kernel 4: out = ctx[2048,256] x Wo[256,512] + bo. Grid (32,8); ~9.25KB LDS.
// =============================================================================
__global__ void __launch_bounds__(256)
outproj_kernel(const float* __restrict__ ctx, const float* __restrict__ Wo,
               const float* __restrict__ bo, float* __restrict__ out)
{
  extern __shared__ char smem[];
  _Float16* at = (_Float16*)smem;            // [64][32]
  _Float16* bt = (_Float16*)(smem + 4096);   // [64 n][BT_STR] (transposed)
  const int m0 = blockIdx.x << 6, n0 = blockIdx.y << 6;
  const int tid = threadIdx.x, lane = tid & 31, wave = tid >> 5;

  v8f c0 = zero8(), c1 = zero8();
  for (int k0 = 0; k0 < INR; k0 += 32) {
    for (int e = tid; e < 64 * 32; e += 256) {
      const int r = e >> 5, kk = e & 31;
      at[e] = (_Float16)ctx[(size_t)(m0 + r) * INR + k0 + kk];
    }
    for (int e = tid; e < 32 * 64; e += 256) {
      const int kk = e >> 6, n = e & 63;
      bt[n * BT_STR + kk] = (_Float16)Wo[(size_t)(k0 + kk) * QD + n0 + n];
    }
    __syncthreads();
#pragma unroll
    for (int t = 0; t < 2; ++t) {
      const int s = wave * 2 + t, mt = s >> 2, nt = s & 3;
      v16h a, bm;
      const int row = mt * 16 + (lane & 15);
#pragma unroll
      for (int hh = 0; hh < 16; ++hh) a[hh] = at[row * 32 + a_kidx(hh, lane)];
      const int n = nt * 16 + (lane & 15);
#pragma unroll
      for (int hh = 0; hh < 16; ++hh) bm[hh] = bt[n * BT_STR + b_kidx(hh, lane)];
      if (t == 0) c0 = wmma32(a, bm, c0);
      else        c1 = wmma32(a, bm, c1);
    }
    __syncthreads();
  }
#pragma unroll
  for (int t = 0; t < 2; ++t) {
    const int s = wave * 2 + t, mt = s >> 2, nt = s & 3;
    const v8f cc = (t == 0) ? c0 : c1;
#pragma unroll
    for (int r = 0; r < 8; ++r) {
      const int row = m0 + mt * 16 + r + ((lane >> 4) << 3);
      const int col = n0 + nt * 16 + (lane & 15);
      out[(size_t)row * QD + col] = cc[r] + bo[col];
    }
  }
}

// =============================================================================
extern "C" void kernel_launch(void* const* d_in, const int* in_sizes, int n_in,
                              void* d_out, int out_size, void* d_ws, size_t ws_size,
                              hipStream_t stream) {
  (void)in_sizes; (void)n_in; (void)out_size; (void)ws_size;

  const float*         x     = (const float*)d_in[0];
  const unsigned char* pm    = (const unsigned char*)d_in[1];
  const unsigned char* vm    = (const unsigned char*)d_in[2];
  const float*         cf    = (const float*)d_in[3];
  const int* db    = (const int*)d_in[4];
  const int* dirb  = (const int*)d_in[5];
  const int* roleb = (const int*)d_in[6];
  const int* hopb  = (const int*)d_in[7];
  const int* edgeb = (const int*)d_in[8];
  const int* spb   = (const int*)d_in[9];
  const int* sameb = (const int*)d_in[10];
  const int* degi  = (const int*)d_in[11];
  const int* cti   = (const int*)d_in[12];
  const float* Wq = (const float*)d_in[13];
  const float* Wk = (const float*)d_in[14];
  const float* Wv = (const float*)d_in[15];
  const float* Wo = (const float*)d_in[16];
  const float* bo = (const float*)d_in[17];
  const float* w1 = (const float*)d_in[18];
  const float* b1 = (const float*)d_in[19];
  const float* w2 = (const float*)d_in[20];
  const float* b2 = (const float*)d_in[21];
  const float* dist_e = (const float*)d_in[22];
  const float* dir_e  = (const float*)d_in[23];
  const float* role_e = (const float*)d_in[24];
  const float* hop_e  = (const float*)d_in[25];
  const float* edge_e = (const float*)d_in[26];
  const float* sp_e   = (const float*)d_in[27];
  const float* deg_e  = (const float*)d_in[28];
  const float* ct_e   = (const float*)d_in[29];
  const float* same_e = (const float*)d_in[30];
  const float* lng    = (const float*)d_in[31];
  const float* lnb    = (const float*)d_in[32];
  const float* outw   = (const float*)d_in[33];
  const float* outb   = (const float*)d_in[34];

  // d_out = out [B,N,QD] ++ attn [B,H,N,N] ++ bias [B,H,N,N]
  float* out_f  = (float*)d_out;
  float* attn_f = out_f + (size_t)BATCH * NTOK * QD;
  float* bias_f = attn_f + (size_t)BATCH * NH * NTOK * NTOK;

  // workspace: qkv f16 (3 x 2048 x 256) then ctx f32 (2048 x 256) ~= 5.25 MB
  _Float16* qkv = (_Float16*)d_ws;
  float* ctx = (float*)((char*)d_ws +
                        (size_t)3 * BATCH * NTOK * INR * sizeof(_Float16));

  pair_bias_kernel<<<dim3(NTOK / 16, NTOK / 16, BATCH), 256, 64 * 1024, stream>>>(
      cf, pm, db, dirb, roleb, hopb, edgeb, spb, sameb, degi, cti,
      w1, b1, w2, b2, dist_e, dir_e, role_e, hop_e, edge_e, sp_e,
      deg_e, ct_e, same_e, lng, lnb, outw, outb, bias_f);

  qkv_kernel<<<dim3((BATCH * NTOK) / 64, INR / 64, 3), 256, 9216 + 128, stream>>>(
      x, Wq, Wk, Wv, qkv);

  const int attn_lds = 65536 + 2048 + 1024 + 8192 + 16 * AT16_STR * 2;
  attn_kernel<<<dim3(NTOK / 16, NH, BATCH), 256, attn_lds, stream>>>(
      qkv, bias_f, pm, vm, attn_f, ctx);

  outproj_kernel<<<dim3((BATCH * NTOK) / 64, QD / 64, 1), 256, 9216 + 128, stream>>>(
      ctx, Wo, bo, out_f);
}
